// ColorHead_4492535792464
// MI455X (gfx1250) — matrix operand order repaired and checked by reading, not compiled, not run on hardware
//
#include <hip/hip_runtime.h>

// MI455X (gfx1250) implementation. wave32. Memory-bound pipeline: everything
// pointwise is fused, stencils are LDS-tiled, the 32->32 1x1 conv is a real
// GEMM via v_wmma_f32_16x16x32_f16, and stencil tiles are filled with the
// CDNA5 direct memory->LDS async path (global_load_async_to_lds_b32 +
// s_wait_asynccnt).

#define EPSF 1e-5f
#define HW     248          // spatial after mode pool
#define PLANE  61504        // 248*248
#define ZPLANE 65536        // 256*256

typedef __attribute__((ext_vector_type(16))) _Float16 v16h;
typedef __attribute__((ext_vector_type(8)))  float    v8f;

__device__ __forceinline__ float lrelu(float x) { return x >= 0.f ? x : 0.01f * x; }

// low 32 bits of a generic pointer into __shared__ = LDS byte offset
__device__ __forceinline__ unsigned lds_off(const void* p) {
  return (unsigned)(unsigned long long)p;
}

// per-lane async copy: 4 bytes, global (saddr + 32-bit byte offset) -> LDS
__device__ __forceinline__ void async_ld_b32(unsigned lds_byte, unsigned gbyte,
                                             const float* base) {
  asm volatile("global_load_async_to_lds_b32 %0, %1, %2"
               :: "v"(lds_byte), "v"(gbyte), "s"(base) : "memory");
}
__device__ __forceinline__ void wait_asynccnt0() {
  asm volatile("s_wait_asynccnt 0" ::: "memory");
}

// ---------------------------------------------------------------- zero init
__global__ void zero_kernel(float* __restrict__ p, int n) {
  int i = blockIdx.x * 256 + threadIdx.x;
  if (i < n) p[i] = 0.f;
}

// ------------------------------------------------- quantize + 11x11 mode pool
// d = round(x*255/16)/16 ; mode over 11x11 (pad=1, zero pad -> level 0),
// tie-break = smallest level (argmax-first). Histogram packed in 2x u64 + int.
__global__ void mode_pool_kernel(const float* __restrict__ x, float* __restrict__ dm) {
  __shared__ unsigned char lev[26 * 26];
  const float* xp = x + (size_t)blockIdx.z * ZPLANE;
  int ox0 = blockIdx.x * 16, oy0 = blockIdx.y * 16;
  int t = threadIdx.y * 16 + threadIdx.x;
  for (int i = t; i < 676; i += 256) {
    int lr = i / 26, lc = i % 26;
    int gy = oy0 - 1 + lr, gx = ox0 - 1 + lc;
    int q = 0;
    if ((unsigned)gy < 256u && (unsigned)gx < 256u)
      q = (int)rintf(xp[gy * 256 + gx] * 15.9375f);   // 255/16 exact; rint = half-even
    lev[i] = (unsigned char)q;
  }
  __syncthreads();
  int oy = oy0 + threadIdx.y, ox = ox0 + threadIdx.x;
  if (oy >= HW || ox >= HW) return;
  unsigned long long a0 = 0ull, a1 = 0ull;
  int c16 = 0;
  #pragma unroll
  for (int ky = 0; ky < 11; ++ky)
    #pragma unroll
    for (int kx = 0; kx < 11; ++kx) {
      int v = lev[(threadIdx.y + ky) * 26 + threadIdx.x + kx];
      if (v < 8)       a0 += 1ull << (v * 8);
      else if (v < 16) a1 += 1ull << ((v - 8) * 8);
      else             c16++;
    }
  int best = (int)(a0 & 0xff), bl = 0;
  #pragma unroll
  for (int l = 1; l < 17; ++l) {
    int c = (l < 8)  ? (int)((a0 >> (l * 8)) & 0xff)
          : (l < 16) ? (int)((a1 >> ((l - 8) * 8)) & 0xff) : c16;
    if (c > best) { best = c; bl = l; }
  }
  dm[(size_t)blockIdx.z * PLANE + oy * HW + ox] = (float)bl * 0.0625f;
}

// ---------------------- conv1 (3->32) + dw scale + BN + LReLU, fully folded
__global__ void conv1_kernel(const float* __restrict__ dm, float* __restrict__ h1,
                             const float* w1, const float* b1, const float* w2, const float* b2,
                             const float* g, const float* bb, const float* m, const float* vv) {
  __shared__ float Wf[96];
  __shared__ float Bf[32];
  int t = threadIdx.x;
  if (t < 32) {
    float s = g[t] * rsqrtf(vv[t] + EPSF);
    float w2t = w2[t];
    Wf[t * 3 + 0] = w1[t * 3 + 0] * w2t * s;
    Wf[t * 3 + 1] = w1[t * 3 + 1] * w2t * s;
    Wf[t * 3 + 2] = w1[t * 3 + 2] * w2t * s;
    Bf[t] = (b1[t] * w2t + b2[t] - m[t]) * s + bb[t];
  }
  __syncthreads();
  int idx = blockIdx.x * 256 + t;            // pixel over 4*61504 (exact grid)
  int b = idx / PLANE, p = idx % PLANE;
  const float* in = dm + (size_t)b * 3 * PLANE + p;
  float v0 = in[0], v1 = in[PLANE], v2 = in[2 * PLANE];
  float* op = h1 + (size_t)b * 32 * PLANE + p;
  #pragma unroll
  for (int o = 0; o < 32; ++o) {
    float y = Wf[o * 3] * v0 + Wf[o * 3 + 1] * v1 + Wf[o * 3 + 2] * v2 + Bf[o];
    op[(size_t)o * PLANE] = lrelu(y);
  }
}

// ----------------- conv2 (32->32) as WMMA GEMM: D = A(16px x 32ch) * B(32 x 16out)
// A layout (16-bit A 16x32, ISA 7.12.2): lane<16 -> M=lane, K in {0..7,16..23};
// lane>=16 -> M=lane-16, K in {8..15,24..31}. B: N = lane&15, same K set per lane as A.
// BN/dw-scale folds hoisted to LDS (one rsqrt per channel per block, not per lane).
__global__ void conv2_wmma_kernel(const float* __restrict__ h1, float* __restrict__ out,
                                  const float* w3, const float* b3, const float* w4, const float* b4,
                                  const float* g, const float* bb, const float* m, const float* vv) {
  __shared__ float sd[8][512];               // per-wave 16px x 32out transpose tile
  __shared__ float wsc[32];                  // w4[o] * bn_scale[o]
  __shared__ float cbs[32];                  // folded output bias
  if (threadIdx.x < 32) {
    int o = threadIdx.x;
    float s = g[o] * rsqrtf(vv[o] + EPSF);
    wsc[o] = w4[o] * s;
    cbs[o] = (b3[o] * w4[o] + b4[o] - m[o]) * s + bb[o];
  }
  __syncthreads();

  int wave = (blockIdx.x * blockDim.x + threadIdx.x) >> 5;
  int lane = threadIdx.x & 31;
  int wv = threadIdx.x >> 5;
  int b = wave / 3844;                       // 61504/16 = 3844 tiles per batch, exact
  int p0 = (wave % 3844) * 16;
  const float* base = h1 + (size_t)b * 32 * PLANE;

  int mrow = lane & 15;
  int kb = (lane >> 4) * 8;
  v16h a;
  #pragma unroll
  for (int e = 0; e < 16; ++e) {
    int k = (e < 8) ? (e + kb) : (e + 8 + kb);
    a[e] = (_Float16)base[(size_t)k * PLANE + p0 + mrow];
  }

  #pragma unroll
  for (int t = 0; t < 2; ++t) {              // two N-tiles of 16 output channels
    v16h bm;
    int c = lane & 15;                       // this lane's N column
    #pragma unroll
    for (int i = 0; i < 16; ++i) {
      int o = t * 16 + c;
      int k = (i < 8) ? (i + kb) : (i + 8 + kb);
      bm[i] = (_Float16)(w3[o * 32 + k] * wsc[o]);
    }
    v8f acc = {};
    acc = __builtin_amdgcn_wmma_f32_16x16x32_f16(false, a, false, bm,
                                                 (short)0, acc, false, false);
    int o = t * 16 + (lane & 15);            // D: N = lane&15 (const across regs)
    float cb = cbs[o];
    #pragma unroll
    for (int r = 0; r < 8; ++r) {            // D: M = r + 8*(lane>>4)
      float val = lrelu(acc[r] + cb);
      int mm = r + ((lane >> 4) << 3);
      sd[wv][mm * 32 + o] = val;
    }
  }
  // LDS ops are in-order within a wave -> coalesced store pass
  float* op = out + (size_t)b * 32 * PLANE;
  #pragma unroll
  for (int j = 0; j < 16; ++j) {
    int o = j * 2 + (lane >> 4);
    int px = lane & 15;
    op[(size_t)o * PLANE + p0 + px] = sd[wv][px * 32 + o];
  }
}

// ------------------------- shared 3x3 conv, stride 3, pad 1 (one kernel, all ch)
__global__ void downgrade_kernel(const float* __restrict__ in, float* __restrict__ out,
                                 int Hin, int Hout, const float* scw, const float* scb) {
  int idx = blockIdx.x * 256 + threadIdx.x;
  int total = 128 * Hout * Hout;
  if (idx >= total) return;
  int ox = idx % Hout, tmp = idx / Hout;
  int oy = tmp % Hout, pl = tmp / Hout;
  const float* ip = in + (size_t)pl * Hin * Hin;
  float acc = scb[0];
  #pragma unroll
  for (int ky = 0; ky < 3; ++ky) {
    int iy = oy * 3 - 1 + ky;
    if ((unsigned)iy >= (unsigned)Hin) continue;
    #pragma unroll
    for (int kx = 0; kx < 3; ++kx) {
      int ix = ox * 3 - 1 + kx;
      if ((unsigned)ix < (unsigned)Hin) acc += scw[ky * 3 + kx] * ip[iy * Hin + ix];
    }
  }
  out[idx] = acc;
}

// --------------------------------- bilinear resize to 256x256 (half-pixel, clamp)
__global__ void resize_kernel(const float* __restrict__ in, float* __restrict__ z, int Hd) {
  int idx = blockIdx.x * 256 + threadIdx.x;        // exact 128*65536
  int ox = idx & 255, oy = (idx >> 8) & 255, pl = idx >> 16;
  const float* ip = in + (size_t)pl * Hd * Hd;
  float scale = (float)Hd * (1.f / 256.f);
  float sy = (oy + 0.5f) * scale - 0.5f;
  float sx = (ox + 0.5f) * scale - 0.5f;
  int y0 = (int)floorf(sy), x0 = (int)floorf(sx);
  float fy = sy - (float)y0, fx = sx - (float)x0;
  int y0c = y0 < 0 ? 0 : (y0 > Hd - 1 ? Hd - 1 : y0);
  int y1 = y0 + 1; int y1c = y1 < 0 ? 0 : (y1 > Hd - 1 ? Hd - 1 : y1);
  int x0c = x0 < 0 ? 0 : (x0 > Hd - 1 ? Hd - 1 : x0);
  int x1 = x0 + 1; int x1c = x1 < 0 ? 0 : (x1 > Hd - 1 ? Hd - 1 : x1);
  float v00 = ip[y0c * Hd + x0c], v01 = ip[y0c * Hd + x1c];
  float v10 = ip[y1c * Hd + x0c], v11 = ip[y1c * Hd + x1c];
  float v0 = v00 + fx * (v01 - v00), v1 = v10 + fx * (v11 - v10);
  z[idx] = v0 + fy * (v1 - v0);
}

// ------- both 5x5 shared-conv branches + BN + LReLU, accumulated into score
// tile fill uses direct memory->LDS async copies (ASYNCcnt path)
__global__ void branch_kernel(const float* __restrict__ z, float* __restrict__ score,
                              const float* aw, const float* ab0, const float* ag,
                              const float* abb, const float* am, const float* av,
                              const float* bw, const float* bb0, const float* bg,
                              const float* bbb, const float* bm, const float* bv) {
  __shared__ float tile[20][20];
  int c = blockIdx.z & 31, b = blockIdx.z >> 5;
  const float* zp = z + (size_t)blockIdx.z * ZPLANE;
  int ox0 = blockIdx.x * 16, oy0 = blockIdx.y * 16;
  int t = threadIdx.y * 16 + threadIdx.x;
  unsigned tbase = lds_off(&tile[0][0]);
  for (int i = t; i < 400; i += 256) {
    int lr = i / 20, lc = i % 20;
    int gy = oy0 - 2 + lr, gx = ox0 - 2 + lc;
    if ((unsigned)gy < 256u && (unsigned)gx < 256u)
      async_ld_b32(tbase + (unsigned)i * 4u, (unsigned)(gy * 256 + gx) * 4u, zp);
    else
      tile[lr][lc] = 0.f;                    // zero padding via plain ds store
  }
  wait_asynccnt0();
  __syncthreads();
  float sa = 0.f, sb = 0.f;
  #pragma unroll
  for (int ky = 0; ky < 5; ++ky)
    #pragma unroll
    for (int kx = 0; kx < 5; ++kx) {
      float p = tile[threadIdx.y + ky][threadIdx.x + kx];
      sa += aw[ky * 5 + kx] * p;
      sb += bw[ky * 5 + kx] * p;
    }
  sa += ab0[0]; sb += bb0[0];
  float s_a = ag[c] * rsqrtf(av[c] + EPSF);
  float s_b = bg[c] * rsqrtf(bv[c] + EPSF);
  sa = lrelu((sa - am[c]) * s_a + abb[c]);
  sb = lrelu((sb - bm[c]) * s_b + bbb[c]);
  int pix = (oy0 + threadIdx.y) * 256 + ox0 + threadIdx.x;
  size_t oa = ((size_t)b * 64 + c) * ZPLANE + pix;
  size_t ob = ((size_t)b * 64 + 32 + c) * ZPLANE + pix;
  score[oa] += sa;
  score[ob] += sb;
}

// ----------------------- ft pointwise: BN -> MaxLeakyReLU(0.7, 0.1) -> BN
__global__ void ft_point_kernel(const float* __restrict__ x, float* __restrict__ y,
                                const float* g1, const float* b1, const float* m1, const float* v1,
                                const float* g2, const float* b2, const float* m2, const float* v2) {
  int idx = blockIdx.x * 256 + threadIdx.x;        // exact 4*64*65536
  int c = (blockIdx.x >> 8) & 63;                  // 256 blocks per channel plane
  float s1 = g1[c] * rsqrtf(v1[c] + EPSF);
  float t = (x[idx] - m1[c]) * s1 + b1[c];
  t = t > 0.1f ? t : 0.7f * t;
  float s2 = g2[c] * rsqrtf(v2[c] + EPSF);
  y[idx] = (t - m2[c]) * s2 + b2[c];
}

// --------------- EmphaseLocal: y * sigmoid(y - 11x11 zero-padded mean /121)
// tile fill via memory->LDS async copies
__global__ void emphase_kernel(const float* __restrict__ y, float* __restrict__ out, float fin) {
  __shared__ float tile[26][26];
  const float* yp = y + (size_t)blockIdx.z * ZPLANE;
  int ox0 = blockIdx.x * 16, oy0 = blockIdx.y * 16;
  int t = threadIdx.y * 16 + threadIdx.x;
  unsigned tbase = lds_off(&tile[0][0]);
  for (int i = t; i < 676; i += 256) {
    int lr = i / 26, lc = i % 26;
    int gy = oy0 - 5 + lr, gx = ox0 - 5 + lc;
    if ((unsigned)gy < 256u && (unsigned)gx < 256u)
      async_ld_b32(tbase + (unsigned)i * 4u, (unsigned)(gy * 256 + gx) * 4u, yp);
    else
      tile[lr][lc] = 0.f;
  }
  wait_asynccnt0();
  __syncthreads();
  float s = 0.f;
  #pragma unroll
  for (int ky = 0; ky < 11; ++ky)
    #pragma unroll
    for (int kx = 0; kx < 11; ++kx)
      s += tile[threadIdx.y + ky][threadIdx.x + kx];
  float yy = tile[threadIdx.y + 5][threadIdx.x + 5];
  float d = yy - s * (1.f / 121.f);
  float gsig = 1.f / (1.f + __expf(-d));
  int pix = (oy0 + threadIdx.y) * 256 + ox0 + threadIdx.x;
  out[(size_t)blockIdx.z * ZPLANE + pix] = yy * gsig * fin;
}

// =============================================================== launcher
extern "C" void kernel_launch(void* const* d_in, const int* in_sizes, int n_in,
                              void* d_out, int out_size, void* d_ws, size_t ws_size,
                              hipStream_t stream) {
  const float* x    = (const float*)d_in[0];
  const float* w1   = (const float*)d_in[1];
  const float* b1   = (const float*)d_in[2];
  const float* w2   = (const float*)d_in[3];
  const float* b2   = (const float*)d_in[4];
  const float* bn1g = (const float*)d_in[5];
  const float* bn1b = (const float*)d_in[6];
  const float* bn1m = (const float*)d_in[7];
  const float* bn1v = (const float*)d_in[8];
  const float* w3   = (const float*)d_in[9];
  const float* b3   = (const float*)d_in[10];
  const float* w4   = (const float*)d_in[11];
  const float* b4   = (const float*)d_in[12];
  const float* bn2g = (const float*)d_in[13];
  const float* bn2b = (const float*)d_in[14];
  const float* bn2m = (const float*)d_in[15];
  const float* bn2v = (const float*)d_in[16];
  const float* scw  = (const float*)d_in[17];
  const float* scb  = (const float*)d_in[18];
  const float* iaw  = (const float*)d_in[19];
  const float* iab0 = (const float*)d_in[20];
  const float* iabg = (const float*)d_in[21];
  const float* iabb = (const float*)d_in[22];
  const float* iabm = (const float*)d_in[23];
  const float* iabv = (const float*)d_in[24];
  const float* ibw  = (const float*)d_in[25];
  const float* ibb0 = (const float*)d_in[26];
  const float* ibbg = (const float*)d_in[27];
  const float* ibbb = (const float*)d_in[28];
  const float* ibbm = (const float*)d_in[29];
  const float* ibbv = (const float*)d_in[30];
  const float* f1g  = (const float*)d_in[31];
  const float* f1b  = (const float*)d_in[32];
  const float* f1m  = (const float*)d_in[33];
  const float* f1v  = (const float*)d_in[34];
  const float* f2g  = (const float*)d_in[35];
  const float* f2b  = (const float*)d_in[36];
  const float* f2m  = (const float*)d_in[37];
  const float* f2v  = (const float*)d_in[38];

  float* ws = (float*)d_ws;
  float* score = (float*)d_out;                 // (4,64,256,256)

  // workspace layout (floats); ft temp aliases the dead front section
  float* dmode = ws;                            // 4*3*248*248   = 738048
  float* h1    = ws + 738048;                   // 4*32*248*248  = 7872512
  float* d0    = ws + 8610560;                  // 4*32*248*248
  const int Hs[6] = {248, 83, 28, 10, 4, 2};
  float* pyr[6];
  pyr[0] = d0;
  size_t off = 16483072;
  for (int i = 1; i < 6; ++i) { pyr[i] = ws + off; off += (size_t)128 * Hs[i] * Hs[i]; }
  float* z    = ws + off;                       // 4*32*256*256 = 8388608  (off = 17480576)
  float* tmpY = ws;                             // 4*64*256*256 aliases [0, z) region

  dim3 b16(16, 16);

  mode_pool_kernel<<<dim3(16, 16, 12), b16, 0, stream>>>(x, dmode);
  conv1_kernel<<<961, 256, 0, stream>>>(dmode, h1, w1, b1, w2, b2, bn1g, bn1b, bn1m, bn1v);
  conv2_wmma_kernel<<<1922, 256, 0, stream>>>(h1, d0, w3, b3, w4, b4, bn2g, bn2b, bn2m, bn2v);
  zero_kernel<<<65536, 256, 0, stream>>>(score, 16777216);

  for (int i = 0; i < 5; ++i) {                 // n = floor(log3(248)) = 5
    int Hi = Hs[i], Ho = Hs[i + 1];
    int tot = 128 * Ho * Ho;
    downgrade_kernel<<<(tot + 255) / 256, 256, 0, stream>>>(pyr[i], pyr[i + 1], Hi, Ho, scw, scb);
    resize_kernel<<<32768, 256, 0, stream>>>(pyr[i + 1], z, Ho);
    branch_kernel<<<dim3(16, 16, 128), b16, 0, stream>>>(z, score,
        iaw, iab0, iabg, iabb, iabm, iabv,
        ibw, ibb0, ibbg, ibbb, ibbm, ibbv);
  }

  for (int j = 0; j < 3; ++j) {
    ft_point_kernel<<<65536, 256, 0, stream>>>(score, tmpY,
        f1g, f1b, f1m, f1v, f2g, f2b, f2m, f2v);
    emphase_kernel<<<dim3(16, 16, 256), b16, 0, stream>>>(tmpY, score,
        (j == 2) ? 0.2f : 1.0f);                // fold final /n into last pass
  }
}
